// GNNModel_3032246911085
// MI455X (gfx1250) — matrix operand order repaired and checked by reading, not compiled, not run on hardware
//
#include <hip/hip_runtime.h>
#include <hip/hip_bf16.h>

// ---------------------------------------------------------------------------
// CDNA5 (gfx1250) 3-layer GCN: deg/rsqrt -> [bf16 WMMA GEMM (packed-B frags)
// -> self-loop init -> edge scatter-add (hw f32 atomics) -> bias+ReLU+softmax] x3
// ---------------------------------------------------------------------------

typedef __attribute__((ext_vector_type(16))) __bf16         v16bf;
typedef __attribute__((ext_vector_type(8)))  float          v8f;
typedef __attribute__((ext_vector_type(8)))  unsigned short v8us;

__device__ __forceinline__ unsigned short f32_to_bf16(float f) {
    unsigned int u = __float_as_uint(f);
    unsigned int r = u + 0x7FFFu + ((u >> 16) & 1u);   // round-to-nearest-even
    return (unsigned short)(r >> 16);
}

// ---------------- degree / norm ----------------
__global__ void k_deg_init(float* deg, int n) {
    int i = blockIdx.x * blockDim.x + threadIdx.x;
    if (i < n) deg[i] = 1.0f;                          // self loop
}

__global__ void k_deg_edges(const int* __restrict__ dst, float* deg, int e) {
    int i = blockIdx.x * blockDim.x + threadIdx.x;
    if (i < e) unsafeAtomicAdd(&deg[dst[i]], 1.0f);    // global_atomic_add_f32
}

__global__ void k_rsqrt(float* deg, int n) {
    int i = blockIdx.x * blockDim.x + threadIdx.x;
    if (i < n) deg[i] = rsqrtf(fmaxf(deg[i], 1.0f));
}

// ---------------- f32 -> bf16 conversion ----------------
__global__ void k_f32_to_bf16(const float* __restrict__ in,
                              unsigned short* __restrict__ out, int n) {
    int i = blockIdx.x * blockDim.x + threadIdx.x;
    if (i < n) out[i] = f32_to_bf16(in[i]);
}

// ---------------- pack W (bf16, [K,DOUT]) into WMMA B-fragment order --------
// Bpack[((tN*(K/32) + kb)*32 + lane)*16 + t] = W[(kb*32 + hi*8 + (t&7) + (t>>3)*16)*DOUT + tN*16 + lo]
// so each GEMM lane reads its 16 halves as 32 contiguous bytes.
__global__ void k_pack_b(const unsigned short* __restrict__ wb,
                         unsigned short* __restrict__ bp, int K, int dout) {
    int idx = blockIdx.x * blockDim.x + threadIdx.x;   // over (dout/16)*(K/32)*32
    int kchunks = K >> 5;
    int total = (dout >> 4) * kchunks * 32;
    if (idx >= total) return;
    int lane = idx & 31;
    int rest = idx >> 5;
    int kb = rest % kchunks;
    int tN = rest / kchunks;
    int lo = lane & 15, hi = lane >> 4;
    unsigned short* out = bp + ((size_t)idx << 4);
    const unsigned short* src = wb + (size_t)(kb * 32 + hi * 8) * dout + tN * 16 + lo;
    #pragma unroll
    for (int t = 0; t < 16; ++t) out[t] = src[(size_t)(t + (t >> 3) * 8) * dout];
}

// ---------------- bf16 WMMA GEMM: C[N,DOUT] = A[N,K] * W[K,DOUT] ------------
// One wave32 computes a 16x64 C strip (4 N-tiles): A frag loaded once per
// K-chunk feeds 4 independent v_wmma_f32_16x16x32_bf16.
template <int K, int DOUT>
__global__ void k_gemm_bf16(const unsigned short* __restrict__ A,
                            const unsigned short* __restrict__ Bp,
                            float* __restrict__ C) {
    constexpr int TN4 = (DOUT / 16) / 4;               // 4 / 2 / 1
    constexpr int KC  = K / 32;                        // 4 or 8
    const int lane = threadIdx.x & 31;
    const int wid  = (blockIdx.x * blockDim.x + threadIdx.x) >> 5;
    const int tM   = wid / TN4;
    const int tN0  = (wid - tM * TN4) * 4;
    const int lo = lane & 15;
    const int hi = lane >> 4;

    const unsigned short* arow = A + (size_t)(tM * 16 + lo) * K;
    v8f acc[4] = {{}, {}, {}, {}};

    #pragma unroll
    for (int kb = 0; kb < KC; ++kb) {
        // A frag (ISA 16-bit A 16x32 layout), two aligned b128 loads
        union { v16bf v; v8us h[2]; } a;
        a.h[0] = *(const v8us*)(arow + kb * 32 + hi * 8);
        a.h[1] = *(const v8us*)(arow + kb * 32 + 16 + hi * 8);
        #pragma unroll
        for (int j = 0; j < 4; ++j) {
            const v8us* bp = (const v8us*)(Bp +
                ((((size_t)(tN0 + j)) * KC + kb) * 32 + lane) * 16);
            union { v16bf v; v8us h[2]; } b;
            b.h[0] = bp[0];
            b.h[1] = bp[1];
            acc[j] = __builtin_amdgcn_wmma_f32_16x16x32_bf16(
                false, a.v, false, b.v, (short)0, acc[j], false, false);
        }
    }

    // C/D layout: VGPR r -> row = tM*16 + r + hi*8, col = tN*16 + lo
    float* crow = C + (size_t)(tM * 16 + hi * 8) * DOUT + lo;
    #pragma unroll
    for (int j = 0; j < 4; ++j)
        #pragma unroll
        for (int r = 0; r < 8; ++r)
            crow[(size_t)r * DOUT + (tN0 + j) * 16] = acc[j][r];
}

// ---------------- aggregation ----------------
// agg[i,:] = h[i,:] * dis[i]^2   (self-loop contribution; initializes agg)
__global__ void k_agg_self(const float* __restrict__ h, const float* __restrict__ dis,
                           float* __restrict__ agg, int n, int dout) {
    int i = blockIdx.x * blockDim.x + threadIdx.x;     // over n*dout
    int node = i / dout;
    float d = dis[node];
    agg[i] = h[i] * d * d;
}

// One wave per edge: agg[dst,:] += h[src,:] * dis[src]*dis[dst]
__global__ void k_agg_edges(const int* __restrict__ src, const int* __restrict__ dst,
                            const float* __restrict__ dis, const float* __restrict__ h,
                            float* __restrict__ agg, int e, int dout) {
    const int lane = threadIdx.x & 31;
    const int edge = blockIdx.x * (blockDim.x >> 5) + (threadIdx.x >> 5);
    if (edge >= e) return;
    const int s = src[edge];
    const int d = dst[edge];
    const float w = dis[s] * dis[d];
    const float* hs = h + (size_t)s * dout;
    float* ad = agg + (size_t)d * dout;
    for (int j = lane; j < dout; j += 32)
        unsafeAtomicAdd(&ad[j], hs[j] * w);            // L2-resident hw fp32 atomics
}

// ---------------- fused bias + ReLU + softmax (one wave32 per row) ----------
__global__ void k_bias_relu_softmax(const float* __restrict__ agg,
                                    const float* __restrict__ bias,
                                    unsigned short* __restrict__ out_bf16,
                                    float* __restrict__ out_f32,
                                    int n, int dout) {
    const int lane = threadIdx.x & 31;
    const int row  = blockIdx.x * (blockDim.x >> 5) + (threadIdx.x >> 5);
    if (row >= n) return;
    const float* a = agg + (size_t)row * dout;
    const int cnt = dout >> 5;                         // 8 / 4 / 2
    float v[8];
    float m = 0.0f;                                    // ReLU => max >= 0
    for (int t = 0; t < cnt; ++t) {
        int j = lane + t * 32;
        float z = fmaxf(a[j] + bias[j], 0.0f);
        v[t] = z;
        m = fmaxf(m, z);
    }
    for (int off = 16; off; off >>= 1) m = fmaxf(m, __shfl_xor(m, off, 32));
    float s = 0.0f;
    for (int t = 0; t < cnt; ++t) { v[t] = __expf(v[t] - m); s += v[t]; }
    for (int off = 16; off; off >>= 1) s += __shfl_xor(s, off, 32);
    const float inv = 1.0f / s;
    for (int t = 0; t < cnt; ++t) {
        int j = lane + t * 32;
        float r = v[t] * inv;
        if (out_bf16) out_bf16[(size_t)row * dout + j] = f32_to_bf16(r);
        else          out_f32 [(size_t)row * dout + j] = r;
    }
}

// ---------------------------------------------------------------------------
extern "C" void kernel_launch(void* const* d_in, const int* in_sizes, int n_in,
                              void* d_out, int out_size, void* d_ws, size_t ws_size,
                              hipStream_t stream) {
    const float* x  = (const float*)d_in[0];
    const int*   ei = (const int*)d_in[1];
    const int E = in_sizes[1] / 2;
    const int N = in_sizes[0] / 128;
    const int* src = ei;
    const int* dst = ei + E;

    const float* W[3] = { (const float*)d_in[2], (const float*)d_in[4], (const float*)d_in[6] };
    const float* b[3] = { (const float*)d_in[3], (const float*)d_in[5], (const float*)d_in[7] };
    const int din[3]  = { 128, 256, 128 };
    const int dout[3] = { 256, 128,  64 };

    char* ws = (char*)d_ws;
    float*          dis   = (float*)ws;                             // 256 KB
    unsigned short* wb    = (unsigned short*)(ws + (256u << 10));   // <=128 KB
    unsigned short* bpack = (unsigned short*)(ws + (512u << 10));   // <=128 KB
    unsigned short* xb    = (unsigned short*)(ws + (1ull  << 20));  // 32 MB
    float*          h     = (float*)(ws + (34ull << 20));           // 64 MB
    float*          agg   = (float*)(ws + (98ull << 20));           // 64 MB

    // degrees -> dis = rsqrt(deg)
    k_deg_init <<<(N + 255) / 256, 256, 0, stream>>>(dis, N);
    k_deg_edges<<<(E + 255) / 256, 256, 0, stream>>>(dst, dis, E);
    k_rsqrt    <<<(N + 255) / 256, 256, 0, stream>>>(dis, N);

    // layer-1 input to bf16
    k_f32_to_bf16<<<(N * 128 + 255) / 256, 256, 0, stream>>>(x, xb, N * 128);

    for (int l = 0; l < 3; ++l) {
        const int K = din[l], M = dout[l];
        const bool last = (l == 2);

        k_f32_to_bf16<<<(K * M + 255) / 256, 256, 0, stream>>>(W[l], wb, K * M);

        const int packThreads = (M / 16) * (K / 32) * 32;
        k_pack_b<<<(packThreads + 255) / 256, 256, 0, stream>>>(wb, bpack, K, M);

        const int waves  = (N / 16) * ((M / 16) / 4);   // 16x64 strip per wave
        const int blocks = waves / 8;                    // 256 threads = 8 waves
        if (l == 0)      k_gemm_bf16<128, 256><<<blocks, 256, 0, stream>>>(xb, bpack, h);
        else if (l == 1) k_gemm_bf16<256, 128><<<blocks, 256, 0, stream>>>(xb, bpack, h);
        else             k_gemm_bf16<128,  64><<<blocks, 256, 0, stream>>>(xb, bpack, h);

        k_agg_self <<<(N * M) / 256, 256, 0, stream>>>(h, dis, agg, N, M);
        k_agg_edges<<<(E + 7) / 8, 256, 0, stream>>>(src, dst, dis, h, agg, E, M);

        k_bias_relu_softmax<<<(N + 7) / 8, 256, 0, stream>>>(
            agg, b[l],
            last ? nullptr : xb,
            last ? (float*)d_out : nullptr,
            N, M);
    }
}
